// FrequencyArtifactExtractor_34041910788295
// MI455X (gfx1250) — matrix operand order repaired and checked by reading, not compiled, not run on hardware
//
#include <hip/hip_runtime.h>
#include <hip/hip_bf16.h>
#include <math.h>

// ---------------------------------------------------------------------------
// FrequencyArtifactExtractor for MI455X (gfx1250, wave32, WMMA)
//   out = concat( haar_dwt2(x), gabor_bank(x, k), fft_channels(x) )
// ---------------------------------------------------------------------------

typedef _Float16 v16h __attribute__((ext_vector_type(16)));
typedef float    v8f  __attribute__((ext_vector_type(8)));

#define GKS     37
#define GPATCH  (GKS * GKS)    // 1369
#define KROW    40             // kernel row padded 37 -> 40 (kills row-wrap + guards)
#define KTOT    1504           // 37*40 = 1480 -> 1504 = 47*32
#define GKSTEPS 47
#define GTILEX  16
#define GTILEY  32
#define INROWS  69             // 31 + 37 + 1 (ky reaches 37 in K pad region)
#define INCOLS  56             // stride; cols used <= 54, 112 B = 28 dwords/row
#define NFILT   18
#define NWROWS  19             // 18 filters + 1 zero row for f>=18 lanes
#define PADW    312            // padded plane stride (624 B, dword multiple)
#define PADH    312
#define PADORG  18

// ---------------------------------------------------------------------------
// Kernel 0a: Gabor weights f32 -> f16, padded-K layout. wsW[f][ky*40+kx].
// ---------------------------------------------------------------------------
__global__ __launch_bounds__(256) void prep_weights_kernel(
    const float* __restrict__ gk, _Float16* __restrict__ wsW)
{
    int i = blockIdx.x * 256 + threadIdx.x;
    if (i >= NWROWS * KTOT) return;
    int f  = i / KTOT;
    int k  = i - f * KTOT;
    int ky = k / KROW;
    int kx = k - ky * KROW;
    float v = (f < NFILT && ky < GKS && kx < GKS)
                  ? gk[f * GPATCH + ky * GKS + kx] : 0.0f;
    wsW[i] = (_Float16)v;
}

// ---------------------------------------------------------------------------
// Kernel 0b: input planes f32 -> f16, zero-padded 312x312 (origin 18,18).
//   Enables pure async-DMA staging of halo tiles in the WMMA kernel.
// ---------------------------------------------------------------------------
__global__ __launch_bounds__(256) void prep_image_kernel(
    const float* __restrict__ x, _Float16* __restrict__ wsXpad)
{
    int i = blockIdx.x * 256 + threadIdx.x;
    if (i >= 24 * PADH * PADW) return;
    int pc    = i % PADW;
    int t     = i / PADW;
    int pr    = t % PADH;
    int plane = t / PADH;
    int gy = pr - PADORG, gx = pc - PADORG;
    float v = (gy >= 0 && gy < 256 && gx >= 0 && gx < 256)
                  ? x[(size_t)plane * 65536 + gy * 256 + gx] : 0.0f;
    wsXpad[i] = (_Float16)v;
}

// ---------------------------------------------------------------------------
// Kernel 1: Haar DWT (memory-bound elementwise)
// ---------------------------------------------------------------------------
__global__ __launch_bounds__(256) void haar_dwt_kernel(
    const float* __restrict__ x, float* __restrict__ out, int total)
{
    int i = blockIdx.x * blockDim.x + threadIdx.x;
    if (i >= total) return;                      // total = B*C*128*128
    int xo = i & 127;
    int t  = i >> 7;
    int yo = t & 127;
    t >>= 7;
    int c = t % 3;
    int b = t / 3;
    const float* p = x + (((size_t)(b * 3 + c) * 256) + 2 * yo) * 256 + 2 * xo;
    float a = p[0], bb = p[1], cc = p[256], d = p[257];
    float LL = (a + bb + cc + d) * 0.5f;
    float LH = (a + bb - cc - d) * 0.5f;
    float HL = (a - bb + cc - d) * 0.5f;
    float HH = (a - bb - cc + d) * 0.5f;
    size_t base = (((size_t)b * 12 + c * 4) * 128 + yo) * 128 + xo;
    out[base]               = LL;
    out[base + 128 * 128]   = LH;
    out[base + 2*128 * 128] = HL;
    out[base + 3*128 * 128] = HH;
}

// ---------------------------------------------------------------------------
// Kernel 2: Gabor bank as implicit GEMM on v_wmma_f32_16x16x32_f16.
//   A (16xK f16)  = weights   (M = filter; LDS row 18 is zeros)
//   B (Kx16 f16)  = patches   (N = pixel column within tile row)
//   Both operands staged via global_load_async_to_lds_b32 (ASYNCcnt),
//   one s_wait_asynccnt + barrier. Per wave: 4 tile rows x 2 filter tiles.
// ---------------------------------------------------------------------------
__global__ __launch_bounds__(256) void gabor_wmma_kernel(
    const _Float16* __restrict__ wsXpad, const _Float16* __restrict__ wsW,
    float* __restrict__ out)
{
    __shared__ __align__(16) unsigned int ldsW32[NWROWS * KTOT / 2]; // 57152 B
    __shared__ __align__(16) _Float16     ldsIn[INROWS * INCOLS];    //  7728 B

    const int tid = threadIdx.x;
    const int bc  = blockIdx.z;
    const int b   = bc / 3;
    const int c   = bc % 3;
    const int tx0 = blockIdx.x * GTILEX;
    const int ty0 = blockIdx.y * GTILEY;

    const _Float16* planeX = wsXpad + (size_t)bc * (PADH * PADW);

    // --- Async DMA 1: weight bank -> LDS (14288 dwords) --------------------
    for (int i = tid; i < NWROWS * KTOT / 2; i += 256) {
        unsigned lds_addr = (unsigned)(uintptr_t)(&ldsW32[i]);
        unsigned voff     = (unsigned)(i * 4);
        asm volatile("global_load_async_to_lds_b32 %0, %1, %2"
                     :: "v"(lds_addr), "v"(voff), "s"(wsW)
                     : "memory");
    }
    // --- Async DMA 2: 69-row halo tile -> LDS (69*28 = 1932 dwords) --------
    {
        const unsigned ldsInBase = (unsigned)(uintptr_t)(&ldsIn[0]);
        for (int i = tid; i < INROWS * (INCOLS / 2); i += 256) {
            int r = i / (INCOLS / 2);
            int d = i - r * (INCOLS / 2);
            unsigned lds_addr = ldsInBase + (unsigned)(i * 4);
            unsigned voff = (unsigned)((ty0 + r) * (PADW * 2) + tx0 * 2 + d * 4);
            asm volatile("global_load_async_to_lds_b32 %0, %1, %2"
                         :: "v"(lds_addr), "v"(voff), "s"(planeX)
                         : "memory");
        }
    }

    asm volatile("s_wait_asynccnt 0x0" ::: "memory");
    __syncthreads();

    const int wave  = tid >> 5;
    const int lane  = tid & 31;
    const int half_ = lane >> 4;   // wave32: lanes 0-15 vs 16-31
    const int ln    = lane & 15;

    const int f0 = ln;                            // M-tile 0 filter row
    const int f1 = (ln < 2) ? (16 + ln) : NFILT;  // M-tile 1: f16,f17 else zero row
    const int wrow0 = f0 * (KTOT / 2);
    const int wrow1 = f1 * (KTOT / 2);

    v8f zero = {0.f, 0.f, 0.f, 0.f, 0.f, 0.f, 0.f, 0.f};
    v8f acc[4][2];                 // [tile-row subtile][filter M-tile]
    #pragma unroll
    for (int i = 0; i < 4; ++i) { acc[i][0] = zero; acc[i][1] = zero; }

    for (int ks = 0; ks < GKSTEPS; ++ks) {
        const int k0 = ks * 32;

        // A fragments (weights), packed-pair b32 loads. 16-bit A layout:
        //   lanes 0-15 : K = {2v+j (v<4), 16+2(v-4)+j (v>=4)};  lanes 16-31: +8
        union { v16h h; unsigned u[8]; } a0, a1;
        #pragma unroll
        for (int v = 0; v < 8; ++v) {
            int off = (v < 4) ? (2 * v) : (16 + 2 * (v - 4));
            int kA  = k0 + off + 8 * half_;       // even
            a0.u[v] = ldsW32[wrow0 + (kA >> 1)];
            a1.u[v] = ldsW32[wrow1 + (kA >> 1)];
        }

        #pragma unroll
        for (int rt = 0; rt < 4; ++rt) {
            const int row = 4 * wave + rt;        // tile row (0..31)
            // B fragment (patches). 16-bit B layout: K = {2v+j (v<4), 16+2(v-4)+j (v>=4)} + 8*half, N = ln.
            // K pad region multiplies zero weights -> no guard needed.
            v16h bfrag;
            #pragma unroll
            for (int v = 0; v < 8; ++v) {
                #pragma unroll
                for (int j = 0; j < 2; ++j) {
                    int k  = k0 + ((v < 4) ? (2 * v) : (16 + 2 * (v - 4))) + j + half_ * 8;
                    int ky = k / KROW;
                    int kx = k - ky * KROW;
                    bfrag[2 * v + j] = ldsIn[(row + ky) * INCOLS + (ln + kx)];
                }
            }
            acc[rt][0] = __builtin_amdgcn_wmma_f32_16x16x32_f16(
                false, a0.h, false, bfrag, (short)0, acc[rt][0], false, false);
            acc[rt][1] = __builtin_amdgcn_wmma_f32_16x16x32_f16(
                false, a1.h, false, bfrag, (short)0, acc[rt][1], false, false);
        }
    }

    // Writeback. D layout: VGPR r -> M = r + 8*half (filter), lane -> N = ln (x).
    // Block-uniform base + 32-bit offsets; filter stride becomes an immediate.
    float* outp = out + ((size_t)b * 54 + (size_t)c * NFILT) * 65536;
    #pragma unroll
    for (int rt = 0; rt < 4; ++rt) {
        const int y    = ty0 + 4 * wave + rt;
        const int xcol = tx0 + ln;
        float* p0 = outp + (half_ * 8) * 65536 + y * 256 + xcol;
        #pragma unroll
        for (int r = 0; r < 8; ++r)               // mt=0: f = r+8*half < 16
            p0[r * 65536] = acc[rt][0][r];
        if (half_ == 0) {                         // mt=1: only f = 16,17 live
            float* p1 = outp + 16 * 65536 + y * 256 + xcol;
            p1[0]     = acc[rt][1][0];
            p1[65536] = acc[rt][1][1];
        }
    }
}

// ---------------------------------------------------------------------------
// Kernel 3a: 256-pt DFT along rows (real -> complex), twiddle-table in LDS.
// ---------------------------------------------------------------------------
__global__ __launch_bounds__(256) void fft_rows_kernel(
    const float* __restrict__ x, float* __restrict__ wsRe, float* __restrict__ wsIm)
{
    __shared__ float sd[256];
    __shared__ float tc[256];
    __shared__ float ts[256];
    const int line = blockIdx.x;          // (b*3+c)*256 + y
    const int t    = threadIdx.x;
    sd[t] = x[(size_t)line * 256 + t];
    float ang = -2.0f * 3.14159265358979323846f * (float)t / 256.0f;
    __sincosf(ang, &ts[t], &tc[t]);       // exp(-2*pi*i*t/256)
    __syncthreads();

    float re = 0.0f, im = 0.0f;
    for (int s = 0; s < 256; ++s) {
        int j   = (t * s) & 255;
        float v = sd[s];
        re += v * tc[j];
        im += v * ts[j];
    }
    wsRe[(size_t)line * 256 + t] = re;
    wsIm[(size_t)line * 256 + t] = im;
}

// ---------------------------------------------------------------------------
// Kernel 3b: 256-pt DFT along columns, ortho scale, mag/phase + zero padding.
// out layout: (B, 6, 256, 256): ch 0..2 magnitude, ch 3..5 phase.
// ---------------------------------------------------------------------------
__global__ __launch_bounds__(256) void fft_cols_kernel(
    const float* __restrict__ wsRe, const float* __restrict__ wsIm,
    float* __restrict__ outFft)
{
    const int xcol = blockIdx.x;          // 0..255
    const int bc   = blockIdx.y;          // 0..23
    const int b    = bc / 3;
    const int c    = bc % 3;
    const int u    = threadIdx.x;

    size_t magIdx = (((size_t)b * 6 + c) * 256 + u) * 256 + xcol;
    size_t phIdx  = (((size_t)b * 6 + 3 + c) * 256 + u) * 256 + xcol;
    if (xcol >= 129) {                    // rfft2 padding region
        outFft[magIdx] = 0.0f;
        outFft[phIdx]  = 0.0f;
        return;                           // uniform per block: safe
    }

    __shared__ float sr[256];
    __shared__ float si[256];
    __shared__ float tc[256];
    __shared__ float ts[256];
    sr[u] = wsRe[((size_t)bc * 256 + u) * 256 + xcol];
    si[u] = wsIm[((size_t)bc * 256 + u) * 256 + xcol];
    float ang = -2.0f * 3.14159265358979323846f * (float)u / 256.0f;
    __sincosf(ang, &ts[u], &tc[u]);
    __syncthreads();

    float re = 0.0f, im = 0.0f;
    for (int y = 0; y < 256; ++y) {
        int j = (u * y) & 255;
        re += sr[y] * tc[j] - si[y] * ts[j];
        im += sr[y] * ts[j] + si[y] * tc[j];
    }
    re *= (1.0f / 256.0f);                // norm='ortho': 1/sqrt(256*256)
    im *= (1.0f / 256.0f);
    outFft[magIdx] = sqrtf(re * re + im * im);
    outFft[phIdx]  = atan2f(im, re);
}

// ---------------------------------------------------------------------------
// Launcher
// ---------------------------------------------------------------------------
extern "C" void kernel_launch(void* const* d_in, const int* in_sizes, int n_in,
                              void* d_out, int out_size, void* d_ws, size_t ws_size,
                              hipStream_t stream)
{
    const float* x  = (const float*)d_in[0];   // (8,3,256,256) f32
    const float* gk = (const float*)d_in[1];   // (18,1,37,37)  f32
    float* out = (float*)d_out;

    const int B = 8, C = 3;
    float* dwt = out;
    float* gab = out + (size_t)B * 12 * 128 * 128;
    float* fft = gab + (size_t)B * 54 * 256 * 256;

    float*     wsRe   = (float*)d_ws;
    float*     wsIm   = wsRe + (size_t)B * C * 256 * 256;
    _Float16*  wsW    = (_Float16*)(wsIm + (size_t)B * C * 256 * 256);
    _Float16*  wsXpad = wsW + (size_t)NWROWS * KTOT;   // dword-aligned

    // 0) f16 prep passes (feed async LDS DMA in gabor kernel)
    int wTotal = NWROWS * KTOT;
    prep_weights_kernel<<<(wTotal + 255) / 256, 256, 0, stream>>>(gk, wsW);
    int iTotal = B * C * PADH * PADW;
    prep_image_kernel<<<(iTotal + 255) / 256, 256, 0, stream>>>(x, wsXpad);

    // 1) Haar DWT
    int dwtTotal = B * C * 128 * 128;
    haar_dwt_kernel<<<(dwtTotal + 255) / 256, 256, 0, stream>>>(x, dwt, dwtTotal);

    // 2) Gabor bank via WMMA implicit GEMM (32x16 spatial tile per block)
    dim3 gg(256 / GTILEX, 256 / GTILEY, B * C);
    gabor_wmma_kernel<<<gg, 256, 0, stream>>>(wsXpad, wsW, gab);

    // 3) rfft2 -> mag/phase
    fft_rows_kernel<<<B * C * 256, 256, 0, stream>>>(x, wsRe, wsIm);
    dim3 gf(256, B * C);
    fft_cols_kernel<<<gf, 256, 0, stream>>>(wsRe, wsIm, fft);
}
